// DotProductAttention_78494822301741
// MI455X (gfx1250) — matrix operand order repaired and checked by reading, not compiled, not run on hardware
//
#include <hip/hip_runtime.h>

// ---------------------------------------------------------------------------
// DotProductAttention for MI455X (gfx1250), wave32 + WMMA bf16.
//   B=32, S=2048, D=64, fp32 I/O.
//   d_out = [ output (B*S*D f32) | attention_weights (B*S*S f32) ]
// One workgroup (4 waves / 128 threads) per 16-query tile => 4096 blocks.
//
// Roofline: mandatory HBM traffic ~554MB stores (weights+out) + ~48MB loads
// => ~26us floor @ 23.3TB/s. bf16 WMMA keeps the 34 GFLOP of matmul far
// below that. Weights/output use non-temporal stores so the 537MB stream
// does not evict the L2-resident K/V (32MB total, re-read ~128x per batch).
// ---------------------------------------------------------------------------

typedef __attribute__((ext_vector_type(16))) __bf16 v16bf;
typedef __attribute__((ext_vector_type(8)))  __bf16 v8bf;
typedef __attribute__((ext_vector_type(8)))  float  v8f;

#define ATT_B   32
#define ATT_S   2048
#define ATT_D   64
#define NKT     (ATT_S / 16)   // 128 score tiles along keys
#define SROW    2052           // padded f32 LDS row stride (breaks bank conflicts)
#define PROW    2056           // padded bf16 LDS row stride (keeps 16B alignment)
#define NEG_INF_VAL (-1000000.0f)
#define NWAVES  4

#define SMEM_SCORE_BYTES (16 * SROW * 4)
#define SMEM_P_BYTES     (16 * PROW * 2)
#define SMEM_RED_BYTES   (16 * 8 * 4)
#define SMEM_BYTES (SMEM_SCORE_BYTES + SMEM_P_BYTES + SMEM_RED_BYTES + 16*4 + 16*4)

__global__ __launch_bounds__(128)
void attn_fused_kernel(const float* __restrict__ Q,
                       const float* __restrict__ K,
                       const float* __restrict__ V,
                       const int*   __restrict__ VL,
                       float* __restrict__ Out,
                       float* __restrict__ W)
{
    extern __shared__ char smem_raw[];
    float*  sS    = (float*)smem_raw;                                  // [16][SROW] f32 scores->exp
    __bf16* sP    = (__bf16*)(smem_raw + SMEM_SCORE_BYTES);            // [16][PROW] bf16 probs
    float*  sRed  = (float*)(smem_raw + SMEM_SCORE_BYTES + SMEM_P_BYTES); // [16][8]
    float*  sMax  = sRed + 16 * 8;                                     // [16]
    float*  sRinv = sMax + 16;                                         // [16]

    const int wg   = blockIdx.x;       // 0..4095
    const int b    = wg >> 7;          // batch
    const int qt   = wg & 127;         // query tile within batch
    const int tid  = threadIdx.x;
    const int wave = tid >> 5;
    const int lane = tid & 31;
    const int hi   = lane >> 4;        // lane half (0/1)
    const int ln   = lane & 15;        // lane-in-half

    const float* Qb = Q + ((size_t)b * ATT_S + (size_t)qt * 16) * ATT_D;
    const float* Kb = K + (size_t)b * ATT_S * ATT_D;
    const float* Vb = V + (size_t)b * ATT_S * ATT_D;
    const int vlen  = VL[b];

    // ---- Load Q A-fragments (bf16), resident for the whole kernel.
    // A 16x32 bf16 layout: lane L holds row M=L%16; element e (v=e/2,p=e%2)
    // maps to K = 2*(v&3) + 16*(v>>2) + 8*(L/16) + p.
    v16bf aq0, aq1;
    {
        const float* qrow = Qb + ln * ATT_D;     // row M = ln
        #pragma unroll
        for (int e = 0; e < 16; ++e) {
            int v = e >> 1, p = e & 1;
            int kk = 2 * (v & 3) + 16 * (v >> 2) + 8 * hi + p;
            aq0[e] = (__bf16)qrow[kk];
            aq1[e] = (__bf16)qrow[32 + kk];
        }
    }

    // =====================================================================
    // Phase 1: scores S = (Q K^T) / 8, masked, staged into LDS (f32).
    // Each wave handles a strided subset of the 128 key tiles.
    // =====================================================================
    for (int kt = wave; kt < NKT; kt += NWAVES) {
        // B 32x16 bf16 layout: lane L holds column N=L%16 (the key index),
        // element e maps to K(=d) = 8*(L/16) + e + 8*(e/8).
        const float* krow = Kb + (size_t)(kt * 16 + ln) * ATT_D;
        v16bf kb0, kb1;
        #pragma unroll
        for (int e = 0; e < 16; ++e) {
            int d0 = 8 * hi + e + 8 * (e >> 3);
            kb0[e] = (__bf16)krow[d0];
            kb1[e] = (__bf16)krow[32 + d0];
        }
        v8f c = {};
        c = __builtin_amdgcn_wmma_f32_16x16x32_bf16(false, aq0, false, kb0,
                                                    (short)0, c, false, false);
        c = __builtin_amdgcn_wmma_f32_16x16x32_bf16(false, aq1, false, kb1,
                                                    (short)0, c, false, false);
        // C/D layout: c[v] -> row M = 8*hi + v, column N = ln.
        const int key   = kt * 16 + ln;
        const bool ok   = key < vlen;               // per-lane select, no branch
        #pragma unroll
        for (int v = 0; v < 8; ++v) {
            float s = ok ? (c[v] * 0.125f) : NEG_INF_VAL;
            sS[(8 * hi + v) * SROW + key] = s;
        }
    }
    __syncthreads();

    // =====================================================================
    // Phase 2: softmax statistics — 8 threads per row. The sum pass writes
    // e = exp(s - m) back in place (thread-private slots, no hazard), so
    // phase 3 only needs a multiply.
    // =====================================================================
    {
        const int row = tid >> 3;     // 0..15
        const int lir = tid & 7;
        float m = -3.0e38f;
        for (int i = lir; i < ATT_S; i += 8)
            m = fmaxf(m, sS[row * SROW + i]);
        sRed[row * 8 + lir] = m;
        __syncthreads();
        if (lir == 0) {
            float mm = sRed[row * 8];
            #pragma unroll
            for (int j = 1; j < 8; ++j) mm = fmaxf(mm, sRed[row * 8 + j]);
            sMax[row] = mm;
        }
        __syncthreads();
        const float mrow = sMax[row];
        float acc = 0.0f;
        for (int i = lir; i < ATT_S; i += 8) {
            float e = __expf(sS[row * SROW + i] - mrow);
            sS[row * SROW + i] = e;                  // in-place: scores -> exp
            acc += e;
        }
        sRed[row * 8 + lir] = acc;
        __syncthreads();
        if (lir == 0) {
            float ss = 0.0f;
            #pragma unroll
            for (int j = 0; j < 8; ++j) ss += sRed[row * 8 + j];
            sRinv[row] = 1.0f / ss;
        }
        __syncthreads();
    }

    // =====================================================================
    // Phase 3: normalize; stream weights to HBM (coalesced, NON-TEMPORAL so
    // the 537MB stream doesn't evict L2-resident K/V) + bf16 P to LDS.
    // =====================================================================
    float* Wb = W + ((size_t)b * ATT_S + (size_t)qt * 16) * ATT_S;
    for (int idx = tid; idx < 16 * ATT_S; idx += 128) {
        const int row = idx >> 11;          // /2048
        const int col = idx & (ATT_S - 1);
        const float p = sS[row * SROW + col] * sRinv[row];
        __builtin_nontemporal_store(p, &Wb[(size_t)row * ATT_S + col]);
        sP[row * PROW + col] = (__bf16)p;
    }
    __syncthreads();

    // =====================================================================
    // Phase 4: output = P @ V. Each wave owns one 16-wide d-slice.
    // =====================================================================
    {
        const int dsl = wave;               // 0..3
        v8f acc = {};
        for (int ks = 0; ks < ATT_S / 32; ++ks) {
            // A-fragment of P from LDS: two contiguous 16B chunks per lane.
            const __bf16* prow = sP + ln * PROW + ks * 32 + 8 * hi;
            v8bf lo  = *(const v8bf*)(prow);        // elements K = ks*32 + 8*hi + 0..7
            v8bf hi8 = *(const v8bf*)(prow + 16);   // elements K = +16
            v16bf ap;
            #pragma unroll
            for (int e = 0; e < 8; ++e) { ap[e] = lo[e]; ap[8 + e] = hi8[e]; }
            // B-fragment of V: lane holds column d = dsl*16 + ln,
            // K(=key) = ks*32 + 8*hi + e + 8*(e/8).
            const float* vcol = Vb + (size_t)(ks * 32 + 8 * hi) * ATT_D + dsl * 16 + ln;
            v16bf bv;
            #pragma unroll
            for (int e = 0; e < 16; ++e) bv[e] = (__bf16)vcol[(size_t)(e + 8 * (e >> 3)) * ATT_D];
            acc = __builtin_amdgcn_wmma_f32_16x16x32_bf16(false, ap, false, bv,
                                                          (short)0, acc, false, false);
        }
        float* orow = Out + ((size_t)b * ATT_S + (size_t)qt * 16) * ATT_D + dsl * 16 + ln;
        #pragma unroll
        for (int v = 0; v < 8; ++v)
            __builtin_nontemporal_store(acc[v], &orow[(size_t)(8 * hi + v) * ATT_D]);
    }
}

extern "C" void kernel_launch(void* const* d_in, const int* in_sizes, int n_in,
                              void* d_out, int out_size, void* d_ws, size_t ws_size,
                              hipStream_t stream) {
    (void)in_sizes; (void)n_in; (void)out_size; (void)d_ws; (void)ws_size;
    const float* Q  = (const float*)d_in[0];
    const float* K  = (const float*)d_in[1];
    const float* V  = (const float*)d_in[2];
    const int*   VL = (const int*)d_in[3];

    float* Out = (float*)d_out;                                   // [B,S,D]
    float* W   = Out + (size_t)ATT_B * ATT_S * ATT_D;             // [B,S,S]

    // >64KB dynamic LDS: raise the limit (no-op each call, not a stream op,
    // safe under graph capture).
    (void)hipFuncSetAttribute((const void*)attn_fused_kernel,
                              hipFuncAttributeMaxDynamicSharedMemorySize,
                              SMEM_BYTES);

    dim3 grid(ATT_B * (ATT_S / 16));   // 4096 workgroups
    dim3 block(128);                   // 4 waves (wave32)
    attn_fused_kernel<<<grid, block, SMEM_BYTES, stream>>>(Q, K, V, VL, Out, W);
}
